// DecoderAttention_41918880808999
// MI455X (gfx1250) — compile-verified, matrix operand order checked by reading, not run on hardware
//
#include <hip/hip_runtime.h>

// ---------------------------------------------------------------------------
// Problem dims (from reference): E=64, D=128, K=4, B=2, H=W=96
// ---------------------------------------------------------------------------
#define Ee   64
#define Dd   128
#define Kc   4
#define Bb   2
#define Hh   96
#define Ww   96
#define Pp   (Hh * Ww)       // 9216 pixels per image
#define NT   (Pp / 16)       // 576 pixel tiles of 16
#define CATC ((Kc + 1) * Ee) // 320 concat channels

typedef __attribute__((ext_vector_type(16))) _Float16 v16h;
typedef __attribute__((ext_vector_type(8)))  _Float16 v8h;
typedef __attribute__((ext_vector_type(8)))  float    v8f;

// ---------------------------------------------------------------------------
// WMMA 16x16x32 f16 fragment loader (row-major f16, leading dim `ld`).
// lane L holds row/col m=L&15; half h=L>>4 selects K sub-blocks:
// elements [0..7] = K = k0 + h*8 + {0..7}, [8..15] = k0 + 16 + h*8 + {0..7}.
// Two 128-bit loads per lane.
// ---------------------------------------------------------------------------
__device__ __forceinline__ v16h load_frag16(const _Float16* base, int ld,
                                            int row0, int k0, int lane) {
  int r = row0 + (lane & 15);
  int h = lane >> 4;
  const _Float16* p = base + (size_t)r * ld + k0 + h * 8;
  v8h lo = *(const v8h*)p;
  v8h hi = *(const v8h*)(p + 16);
  v16h f;
#pragma unroll
  for (int i = 0; i < 8; ++i) { f[i] = lo[i]; f[i + 8] = hi[i]; }
  return f;
}

// ---------------------------------------------------------------------------
// Conversion / transpose kernels (f32 NCHW -> f16 pixel-major, weights -> f16)
// ---------------------------------------------------------------------------
__global__ void cvt_xq(const float* __restrict__ dec, _Float16* __restrict__ xqT) {
  int idx = blockIdx.x * blockDim.x + threadIdx.x;            // (b*P + p)*D + d
  if (idx >= Bb * Pp * Dd) return;
  int d = idx % Dd;
  int bp = idx / Dd;
  int p = bp % Pp;
  int b = bp / Pp;
  xqT[idx] = (_Float16)dec[((size_t)b * Dd + d) * Pp + p];
}

__global__ void cvt_xc(const float* __restrict__ ctx, _Float16* __restrict__ xcT) {
  int idx = blockIdx.x * blockDim.x + threadIdx.x;            // (kb*P + p)*E + e
  if (idx >= Kc * Bb * Pp * Ee) return;
  int e = idx % Ee;
  int t = idx / Ee;
  int p = t % Pp;
  int kb = t / Pp;
  xcT[idx] = (_Float16)ctx[((size_t)kb * Ee + e) * Pp + p];
}

__global__ void cvt_w(const float* __restrict__ wdec, const float* __restrict__ wenc,
                      const float* __restrict__ wattn,
                      _Float16* __restrict__ Wdec, _Float16* __restrict__ Wenc,
                      _Float16* __restrict__ Wattn) {
  int idx = blockIdx.x * blockDim.x + threadIdx.x;
  const int n_dec = Ee * Dd;           // 8192
  const int n_enc = Ee * Ee;           // 4096
  const int n_att = Ee * CATC;         // 20480
  if (idx < n_dec) {
    Wdec[idx] = (_Float16)wdec[idx];
  } else if (idx < n_dec + n_enc) {
    Wenc[idx - n_dec] = (_Float16)wenc[idx - n_dec];
  } else if (idx < n_dec + n_enc + n_att) {
    Wattn[idx - n_dec - n_enc] = (_Float16)wattn[idx - n_dec - n_enc];
  }
}

__global__ void cvt_wv9(const float* __restrict__ wval, _Float16* __restrict__ Wv9) {
  int idx = blockIdx.x * blockDim.x + threadIdx.x;            // t*(E*D) + o*D + i
  if (idx >= 9 * Ee * Dd) return;
  int t = idx / (Ee * Dd);
  int rem = idx % (Ee * Dd);                                  // o*D + i
  Wv9[idx] = (_Float16)wval[(size_t)rem * 9 + t];             // w_val[o][i][r][c]
}

// ---------------------------------------------------------------------------
// queries = w_dec(64x128) @ decoded + b_dec  ->  qT f32 [b][p][e]
// one wave per 16-pixel tile computes all 64 output channels:
// B fragment loaded once per K-step, reused by 4 WMMAs (4 accumulators).
// ---------------------------------------------------------------------------
__global__ void queries_gemm(const _Float16* __restrict__ Wdec,
                             const _Float16* __restrict__ xqT,
                             const float* __restrict__ b_dec,
                             float* __restrict__ qT, int total) {
  int lane = threadIdx.x & 31;
  int gw = blockIdx.x * (blockDim.x >> 5) + (threadIdx.x >> 5);
  if (gw >= total) return;
  int nt = gw % NT;
  int b = gw / NT;
  const _Float16* X = xqT + (size_t)b * Pp * Dd;
  __builtin_prefetch(X + (size_t)nt * 16 * Dd, 0, 1);         // global_prefetch_b8
  v8f acc0 = {}, acc1 = {}, acc2 = {}, acc3 = {};
#pragma unroll
  for (int k0 = 0; k0 < Dd; k0 += 32) {
    v16h bf = load_frag16(X, Dd, nt * 16, k0, lane);
    v16h a0 = load_frag16(Wdec, Dd, 0,  k0, lane);
    v16h a1 = load_frag16(Wdec, Dd, 16, k0, lane);
    v16h a2 = load_frag16(Wdec, Dd, 32, k0, lane);
    v16h a3 = load_frag16(Wdec, Dd, 48, k0, lane);
    acc0 = __builtin_amdgcn_wmma_f32_16x16x32_f16(false, a0, false, bf, (short)0, acc0, false, false);
    acc1 = __builtin_amdgcn_wmma_f32_16x16x32_f16(false, a1, false, bf, (short)0, acc1, false, false);
    acc2 = __builtin_amdgcn_wmma_f32_16x16x32_f16(false, a2, false, bf, (short)0, acc2, false, false);
    acc3 = __builtin_amdgcn_wmma_f32_16x16x32_f16(false, a3, false, bf, (short)0, acc3, false, false);
  }
  int n  = nt * 16 + (lane & 15);
  int hb = (lane >> 4) * 8;
  float* out = qT + ((size_t)(b * Pp + n)) * Ee;
  v8f accs[4] = {acc0, acc1, acc2, acc3};
#pragma unroll
  for (int mt = 0; mt < 4; ++mt)
#pragma unroll
    for (int j = 0; j < 8; ++j) {
      int m = mt * 16 + hb + j;
      out[m] = accs[mt][j] + b_dec[m];
    }
}

// ---------------------------------------------------------------------------
// keys[k] = w_enc(64x64) @ ctx[k] + b_enc  ->  keysT f32 [k][b][p][e]
// ---------------------------------------------------------------------------
__global__ void keys_gemm(const _Float16* __restrict__ Wenc,
                          const _Float16* __restrict__ xcT,
                          const float* __restrict__ b_enc,
                          float* __restrict__ keysT, int total) {
  int lane = threadIdx.x & 31;
  int gw = blockIdx.x * (blockDim.x >> 5) + (threadIdx.x >> 5);
  if (gw >= total) return;
  int nt = gw % NT;
  int kb = gw / NT;                     // k*B + b
  const _Float16* X = xcT + (size_t)kb * Pp * Ee;
  __builtin_prefetch(X + (size_t)nt * 16 * Ee, 0, 1);
  v8f acc0 = {}, acc1 = {}, acc2 = {}, acc3 = {};
#pragma unroll
  for (int k0 = 0; k0 < Ee; k0 += 32) {
    v16h bf = load_frag16(X, Ee, nt * 16, k0, lane);
    v16h a0 = load_frag16(Wenc, Ee, 0,  k0, lane);
    v16h a1 = load_frag16(Wenc, Ee, 16, k0, lane);
    v16h a2 = load_frag16(Wenc, Ee, 32, k0, lane);
    v16h a3 = load_frag16(Wenc, Ee, 48, k0, lane);
    acc0 = __builtin_amdgcn_wmma_f32_16x16x32_f16(false, a0, false, bf, (short)0, acc0, false, false);
    acc1 = __builtin_amdgcn_wmma_f32_16x16x32_f16(false, a1, false, bf, (short)0, acc1, false, false);
    acc2 = __builtin_amdgcn_wmma_f32_16x16x32_f16(false, a2, false, bf, (short)0, acc2, false, false);
    acc3 = __builtin_amdgcn_wmma_f32_16x16x32_f16(false, a3, false, bf, (short)0, acc3, false, false);
  }
  int n  = nt * 16 + (lane & 15);
  int hb = (lane >> 4) * 8;
  float* out = keysT + ((size_t)kb * Pp + n) * Ee;
  v8f accs[4] = {acc0, acc1, acc2, acc3};
#pragma unroll
  for (int mt = 0; mt < 4; ++mt)
#pragma unroll
    for (int j = 0; j < 8; ++j) {
      int m = mt * 16 + hb + j;
      out[m] = accs[mt][j] + b_enc[m];
    }
}

// ---------------------------------------------------------------------------
// values = conv3x3(decoded, w_val) + b_val  ->  catT[:, :, 0:64] (f16)
// implicit im2col: 9 shifted GEMMs with per-lane border zero-fill.
// The (expensive) shifted B fragment is loaded once per (tap, K-step) and
// reused across all 4 output-channel tiles: 144 WMMA per wave.
// ---------------------------------------------------------------------------
__global__ void values_gemm(const _Float16* __restrict__ Wv9,
                            const _Float16* __restrict__ xqT,
                            const float* __restrict__ b_val,
                            _Float16* __restrict__ catT, int total) {
  int lane = threadIdx.x & 31;
  int gw = blockIdx.x * (blockDim.x >> 5) + (threadIdx.x >> 5);
  if (gw >= total) return;
  int nt = gw % NT;
  int b = gw / NT;
  int y = nt / (Ww / 16);
  int x0 = (nt % (Ww / 16)) * 16;
  const _Float16* X = xqT + (size_t)b * Pp * Dd;
  int nlane = lane & 15;
  int h = lane >> 4;
  v8f acc0 = {}, acc1 = {}, acc2 = {}, acc3 = {};
  for (int t9 = 0; t9 < 9; ++t9) {
    int dy = t9 / 3 - 1, dx = t9 % 3 - 1;
    int yy = y + dy;
    int xx = x0 + nlane + dx;
    bool valid = (yy >= 0) && (yy < Hh) && (xx >= 0) && (xx < Ww);
    const _Float16* Xp = valid ? (X + (size_t)(yy * Ww + xx) * Dd) : X;
    const _Float16* Wt = Wv9 + (size_t)t9 * Ee * Dd;
#pragma unroll
    for (int k0 = 0; k0 < Dd; k0 += 32) {
      v16h bf;
#pragma unroll
      for (int i = 0; i < 16; ++i) bf[i] = (_Float16)0.0f;
      if (valid) {
        const _Float16* p = Xp + k0 + h * 8;
        v8h lo = *(const v8h*)p;
        v8h hi = *(const v8h*)(p + 16);
#pragma unroll
        for (int i = 0; i < 8; ++i) { bf[i] = lo[i]; bf[i + 8] = hi[i]; }
      }
      v16h a0 = load_frag16(Wt, Dd, 0,  k0, lane);
      v16h a1 = load_frag16(Wt, Dd, 16, k0, lane);
      v16h a2 = load_frag16(Wt, Dd, 32, k0, lane);
      v16h a3 = load_frag16(Wt, Dd, 48, k0, lane);
      acc0 = __builtin_amdgcn_wmma_f32_16x16x32_f16(false, a0, false, bf, (short)0, acc0, false, false);
      acc1 = __builtin_amdgcn_wmma_f32_16x16x32_f16(false, a1, false, bf, (short)0, acc1, false, false);
      acc2 = __builtin_amdgcn_wmma_f32_16x16x32_f16(false, a2, false, bf, (short)0, acc2, false, false);
      acc3 = __builtin_amdgcn_wmma_f32_16x16x32_f16(false, a3, false, bf, (short)0, acc3, false, false);
    }
  }
  int n = nt * 16 + nlane;
  _Float16* out = catT + ((size_t)(b * Pp + n)) * CATC;       // channels 0..63
  v8f accs[4] = {acc0, acc1, acc2, acc3};
#pragma unroll
  for (int mt = 0; mt < 4; ++mt)
#pragma unroll
    for (int j = 0; j < 8; ++j) {
      int m = mt * 16 + h * 8 + j;
      out[m] = (_Float16)(accs[mt][j] + b_val[m]);
    }
}

// ---------------------------------------------------------------------------
// neighborhood attention: one wave per (k, b, pixel); 2 channels per lane.
// score_n = sum_e w_agg[e]*tanh(q[e] + key_n[e]); softmax over 9 (+border
// mask); attended[e] = sum_n p_n * ctx_n[e]  ->  catT channels 64*(k+1)..
// b_agg cancels in softmax and is omitted.
// ---------------------------------------------------------------------------
__global__ void attn_kernel(const float* __restrict__ qT,
                            const float* __restrict__ keysT,
                            const _Float16* __restrict__ xcT,
                            const float* __restrict__ w_agg,
                            _Float16* __restrict__ catT, int total) {
  int lane = threadIdx.x & 31;
  int gw = blockIdx.x * (blockDim.x >> 5) + (threadIdx.x >> 5);
  if (gw >= total) return;
  int p = gw % Pp;
  int t = gw / Pp;
  int b = t % Bb;
  int k = t / Bb;
  int y = p / Ww, x = p % Ww;
  int e0 = lane * 2;

  const float* qrow = qT + ((size_t)(b * Pp + p)) * Ee;
  float q0 = qrow[e0], q1 = qrow[e0 + 1];
  float wa0 = w_agg[e0], wa1 = w_agg[e0 + 1];

  const float*    keyb = keysT + ((size_t)(k * Bb + b)) * Pp * Ee;
  const _Float16* ctxb = xcT  + ((size_t)(k * Bb + b)) * Pp * Ee;

  float s[9];
  int   pp[9];
  bool  val[9];
#pragma unroll
  for (int n = 0; n < 9; ++n) {
    int dy = n / 3 - 1, dx = n % 3 - 1;
    int yy = y + dy, xx = x + dx;
    bool v = (yy >= 0) && (yy < Hh) && (xx >= 0) && (xx < Ww);   // wave-uniform
    val[n] = v;
    int pn = v ? (yy * Ww + xx) : 0;
    pp[n] = pn;
    float partial = 0.0f;
    if (v) {
      float k0v = keyb[(size_t)pn * Ee + e0];
      float k1v = keyb[(size_t)pn * Ee + e0 + 1];
      partial = wa0 * tanhf(q0 + k0v) + wa1 * tanhf(q1 + k1v);
    }
#pragma unroll
    for (int m = 16; m >= 1; m >>= 1) partial += __shfl_xor(partial, m, 32);
    s[n] = v ? partial : -1e9f;
  }

  float mx = s[0];
#pragma unroll
  for (int n = 1; n < 9; ++n) mx = fmaxf(mx, s[n]);
  float pr[9];
  float sum = 0.0f;
#pragma unroll
  for (int n = 0; n < 9; ++n) { pr[n] = __expf(s[n] - mx); sum += pr[n]; }
  float inv = 1.0f / sum;

  float a0 = 0.0f, a1 = 0.0f;
#pragma unroll
  for (int n = 0; n < 9; ++n) {
    if (!val[n]) continue;                                      // wave-uniform
    float pn = pr[n] * inv;
    float c0 = (float)ctxb[(size_t)pp[n] * Ee + e0];
    float c1 = (float)ctxb[(size_t)pp[n] * Ee + e0 + 1];
    a0 += pn * c0;
    a1 += pn * c1;
  }
  _Float16* out = catT + ((size_t)(b * Pp + p)) * CATC + Ee * (k + 1) + e0;
  out[0] = (_Float16)a0;
  out[1] = (_Float16)a1;
}

// ---------------------------------------------------------------------------
// out = LeakyReLU_0.2( w_attn(64x320) @ cat + b_attn )  -> d_out [B,E,H,W] f32
// K = 320 -> 10 K-steps; one wave per 16-pixel tile, all 64 out channels.
// ---------------------------------------------------------------------------
__global__ void final_gemm(const _Float16* __restrict__ Wattn,
                           const _Float16* __restrict__ catT,
                           const float* __restrict__ b_attn,
                           float* __restrict__ out, int total) {
  int lane = threadIdx.x & 31;
  int gw = blockIdx.x * (blockDim.x >> 5) + (threadIdx.x >> 5);
  if (gw >= total) return;
  int nt = gw % NT;
  int b = gw / NT;
  const _Float16* X = catT + (size_t)b * Pp * CATC;
  __builtin_prefetch(X + (size_t)nt * 16 * CATC, 0, 1);
  v8f acc0 = {}, acc1 = {}, acc2 = {}, acc3 = {};
#pragma unroll
  for (int k0 = 0; k0 < CATC; k0 += 32) {
    v16h bf = load_frag16(X, CATC, nt * 16, k0, lane);
    v16h a0 = load_frag16(Wattn, CATC, 0,  k0, lane);
    v16h a1 = load_frag16(Wattn, CATC, 16, k0, lane);
    v16h a2 = load_frag16(Wattn, CATC, 32, k0, lane);
    v16h a3 = load_frag16(Wattn, CATC, 48, k0, lane);
    acc0 = __builtin_amdgcn_wmma_f32_16x16x32_f16(false, a0, false, bf, (short)0, acc0, false, false);
    acc1 = __builtin_amdgcn_wmma_f32_16x16x32_f16(false, a1, false, bf, (short)0, acc1, false, false);
    acc2 = __builtin_amdgcn_wmma_f32_16x16x32_f16(false, a2, false, bf, (short)0, acc2, false, false);
    acc3 = __builtin_amdgcn_wmma_f32_16x16x32_f16(false, a3, false, bf, (short)0, acc3, false, false);
  }
  int n  = nt * 16 + (lane & 15);
  int hb = (lane >> 4) * 8;
  v8f accs[4] = {acc0, acc1, acc2, acc3};
#pragma unroll
  for (int mt = 0; mt < 4; ++mt)
#pragma unroll
    for (int j = 0; j < 8; ++j) {
      int m = mt * 16 + hb + j;
      float v = accs[mt][j] + b_attn[m];
      out[((size_t)b * Ee + m) * Pp + n] = (v >= 0.0f) ? v : 0.2f * v;
    }
}

// ---------------------------------------------------------------------------
// launch
// ---------------------------------------------------------------------------
extern "C" void kernel_launch(void* const* d_in, const int* in_sizes, int n_in,
                              void* d_out, int out_size, void* d_ws, size_t ws_size,
                              hipStream_t stream) {
  const float* contexts = (const float*)d_in[0];   // [K,B,E,H,W]
  const float* decoded  = (const float*)d_in[1];   // [B,D,H,W]
  const float* w_enc    = (const float*)d_in[2];   // [E,E]
  const float* b_enc    = (const float*)d_in[3];   // [E]
  const float* w_dec    = (const float*)d_in[4];   // [E,D]
  const float* b_dec    = (const float*)d_in[5];   // [E]
  const float* w_agg    = (const float*)d_in[6];   // [E]
  // d_in[7] = b_agg (scalar): cancels in softmax, unused
  const float* w_val    = (const float*)d_in[8];   // [E,D,3,3]
  const float* b_val    = (const float*)d_in[9];   // [E]
  const float* w_attn   = (const float*)d_in[10];  // [E,(K+1)*E]
  const float* b_attn   = (const float*)d_in[11];  // [E]
  float* out = (float*)d_out;

  // workspace carve-up (bytes); total ~49.8 MB
  char* ws = (char*)d_ws;
  size_t off = 0;
  _Float16* xqT   = (_Float16*)(ws + off); off += (size_t)Bb * Pp * Dd * 2;        // 4.72 MB
  _Float16* xcT   = (_Float16*)(ws + off); off += (size_t)Kc * Bb * Pp * Ee * 2;   // 9.44 MB
  _Float16* Wdec  = (_Float16*)(ws + off); off += (size_t)Ee * Dd * 2;
  _Float16* Wenc  = (_Float16*)(ws + off); off += (size_t)Ee * Ee * 2;
  _Float16* Wattn = (_Float16*)(ws + off); off += (size_t)Ee * CATC * 2;
  _Float16* Wv9   = (_Float16*)(ws + off); off += (size_t)9 * Ee * Dd * 2;
  float*    qT    = (float*)(ws + off);    off += (size_t)Bb * Pp * Ee * 4;        // 4.72 MB
  float*    keysT = (float*)(ws + off);    off += (size_t)Kc * Bb * Pp * Ee * 4;   // 18.9 MB
  _Float16* catT  = (_Float16*)(ws + off); off += (size_t)Bb * Pp * CATC * 2;      // 11.8 MB

  const int TB = 256;
  // conversions
  {
    int n = Bb * Pp * Dd;
    cvt_xq<<<(n + TB - 1) / TB, TB, 0, stream>>>(decoded, xqT);
  }
  {
    int n = Kc * Bb * Pp * Ee;
    cvt_xc<<<(n + TB - 1) / TB, TB, 0, stream>>>(contexts, xcT);
  }
  {
    int n = Ee * Dd + Ee * Ee + Ee * CATC;
    cvt_w<<<(n + TB - 1) / TB, TB, 0, stream>>>(w_dec, w_enc, w_attn, Wdec, Wenc, Wattn);
  }
  {
    int n = 9 * Ee * Dd;
    cvt_wv9<<<(n + TB - 1) / TB, TB, 0, stream>>>(w_val, Wv9);
  }

  const int GB = 128;                  // 4 waves / block
  // queries GEMM: one wave per pixel tile (B * 576)
  {
    int total = Bb * NT;
    queries_gemm<<<(total + 3) / 4, GB, 0, stream>>>(Wdec, xqT, b_dec, qT, total);
  }
  // keys GEMM: K*B * 576 tiles
  {
    int total = Kc * Bb * NT;
    keys_gemm<<<(total + 3) / 4, GB, 0, stream>>>(Wenc, xcT, b_enc, keysT, total);
  }
  // values conv-GEMM: B * 576 tiles
  {
    int total = Bb * NT;
    values_gemm<<<(total + 3) / 4, GB, 0, stream>>>(Wv9, xqT, b_val, catT, total);
  }
  // attention: K*B*P waves
  {
    int total = Kc * Bb * Pp;
    attn_kernel<<<(total + 3) / 4, GB, 0, stream>>>(qT, keysT, xcT, w_agg, catT, total);
  }
  // final GEMM + LeakyReLU
  {
    int total = Bb * NT;
    final_gemm<<<(total + 3) / 4, GB, 0, stream>>>(Wattn, catT, b_attn, out, total);
  }
  (void)in_sizes; (void)n_in; (void)out_size; (void)ws_size;
}
